// CausalMultiheadSelfAttention_31121333027322
// MI455X (gfx1250) — matrix operand order repaired and checked by reading, not compiled, not run on hardware
//
#include <hip/hip_runtime.h>
#include <hip/hip_bf16.h>
#include <math.h>

typedef __bf16 bf16;
typedef bf16  v16bf __attribute__((ext_vector_type(16)));
typedef bf16  v8bf  __attribute__((ext_vector_type(8)));
typedef float v8f   __attribute__((ext_vector_type(8)));

constexpr int Bc  = 2;
constexpr int Tc  = 2048;
constexpr int Dc  = 1024;
constexpr int Hc  = 16;
constexpr int HDc = 64;

// DPP16 ROW_XMASK cross-lane (stays within each 16-lane DPP row, which matches
// the lane-half row ownership of the WMMA C/D layout).
template <int MASK>
__device__ inline float dpp_xmask_f(float x) {
    int i = __float_as_int(x);
    int y = __builtin_amdgcn_update_dpp(i, i, 0x160 | MASK, 0xf, 0xf, true);
    return __int_as_float(y);
}

__device__ inline float rowhalf_max(float x) {
    x = fmaxf(x, dpp_xmask_f<1>(x));
    x = fmaxf(x, dpp_xmask_f<2>(x));
    x = fmaxf(x, dpp_xmask_f<4>(x));
    x = fmaxf(x, dpp_xmask_f<8>(x));
    return x;
}

// Load a 16-bit A fragment (16x32 layout): elem j -> k=k0+hi*16+j,
// from a row-major row pointer.
__device__ inline v16bf load_a_frag(const bf16* rowp, int k0, int hi) {
    const bf16* p0 = rowp + k0 + hi * 16;
    v8bf lo = *(const v8bf*)p0;
    v8bf hh = *(const v8bf*)(p0 + 8);
    v16bf a;
#pragma unroll
    for (int i = 0; i < 8; ++i) { a[i] = lo[i]; a[i + 8] = hh[i]; }
    return a;
}

// ---------------------------------------------------------------------------
// f32 -> bf16 conversion
// ---------------------------------------------------------------------------
__global__ void cvt_f32_bf16(const float* __restrict__ src, bf16* __restrict__ dst, int n) {
    int i = blockIdx.x * blockDim.x + threadIdx.x;
    if (i < n) dst[i] = (bf16)src[i];
}

// ---------------------------------------------------------------------------
// GEMM: C[m][n] = sum_k X[m][k] * W[n][k] + bias[n]
// X: (M,K) bf16 row-major.  W: (N,K) bf16 row-major.
// MODE 0: store bf16 row-major dst[m*ld2+n]
// MODE 1: store bf16 per-batch transposed: m=(bb*ld2+tt) -> dst[(bb*N+n)*ld2+tt]
// MODE 2: store f32 row-major dst[m*ld2+n]
// One wave computes a 32x64 tile: 2 A fragments x 4 B fragments = 8 WMMAs/k-step.
// ---------------------------------------------------------------------------
template <int MODE>
__global__ void gemm_xwT(const bf16* __restrict__ X, const bf16* __restrict__ W,
                         const float* __restrict__ bias, void* __restrict__ dstv,
                         int M, int N, int K, int ld2) {
    const int lane = threadIdx.x & 31;
    const int lrow = lane & 15;
    const int hi   = lane >> 4;
    const int w    = blockIdx.x * (blockDim.x >> 5) + (threadIdx.x >> 5);
    const int n64  = N >> 6;
    const int mt   = w / n64;
    const int nt   = w - mt * n64;
    if (mt * 32 >= M) return;
    const int m0 = mt * 32, n0 = nt * 64;

    const bf16* xrow0 = X + (size_t)(m0 + lrow) * K;
    const bf16* xrow1 = X + (size_t)(m0 + 16 + lrow) * K;
    v8f acc[2][4] = {};

    for (int kk = 0; kk < K; kk += 32) {
        v16bf a0 = load_a_frag(xrow0, kk, hi);
        v16bf a1 = load_a_frag(xrow1, kk, hi);
#pragma unroll
        for (int t = 0; t < 4; ++t) {
            // B fragment: lane holds n = n0+t*16+lrow (row of W), k = kk + 16*hi + j (contig)
            const bf16* wr = W + (size_t)(n0 + t * 16 + lrow) * K + kk + hi * 16;
            v16bf bfr = *(const v16bf*)wr;
            acc[0][t] = __builtin_amdgcn_wmma_f32_16x16x32_bf16(
                false, a0, false, bfr, (short)0, acc[0][t], false, false);
            acc[1][t] = __builtin_amdgcn_wmma_f32_16x16x32_bf16(
                false, a1, false, bfr, (short)0, acc[1][t], false, false);
        }
    }

#pragma unroll
    for (int u = 0; u < 2; ++u)
#pragma unroll
        for (int t = 0; t < 4; ++t)
#pragma unroll
            for (int r = 0; r < 8; ++r) {
                int m = m0 + u * 16 + r + 8 * hi;
                int n = n0 + t * 16 + lrow;
                float v = acc[u][t][r] + bias[n];
                if (MODE == 0) {
                    ((bf16*)dstv)[(size_t)m * ld2 + n] = (bf16)v;
                } else if (MODE == 1) {
                    int bb = m / ld2, tt = m - bb * ld2;
                    ((bf16*)dstv)[((size_t)bb * N + n) * ld2 + tt] = (bf16)v;
                } else {
                    ((float*)dstv)[(size_t)m * ld2 + n] = v;
                }
            }
}

// ---------------------------------------------------------------------------
// RoPE, applied in place on bf16 Q and K (layout B,T,D). One thread per pair.
// ---------------------------------------------------------------------------
__global__ void rope_kernel(bf16* __restrict__ q, bf16* __restrict__ k,
                            const int* __restrict__ pos) {
    int tid = blockIdx.x * blockDim.x + threadIdx.x;
    int i   = tid & 31;          // pair index within head dim (half = 32)
    int hh  = (tid >> 5) & 15;   // head
    int ft  = tid >> 9;          // b*T + t
    if (ft >= Bc * Tc) return;
    int t = ft & (Tc - 1);
    float p = (float)pos[t];
    float freq = __expf(-((float)(2 * i) / (float)HDc) * logf(10000.0f));
    float s, c;
    __sincosf(p * freq, &s, &c);
    size_t base = (size_t)ft * Dc + hh * HDc + 2 * i;
    float q1 = (float)q[base], q2 = (float)q[base + 1];
    q[base]     = (bf16)(q1 * c - q2 * s);
    q[base + 1] = (bf16)(q1 * s + q2 * c);
    float k1 = (float)k[base], k2 = (float)k[base + 1];
    k[base]     = (bf16)(k1 * c - k2 * s);
    k[base + 1] = (bf16)(k1 * s + k2 * c);
}

// ---------------------------------------------------------------------------
// Flash attention: one wave per 16-query tile per (b,h).
// Q,K: (B,T,D) bf16 (RoPE'd).  Vt: (B,D,T) bf16.  att out: (B,T,D) bf16.
// Keys streamed in chunks of 32 with online softmax.
//  - row max: DPP row_xmask butterfly (pure VALU, no LDS)
//  - row sum: extra WMMA  P @ ones  into a running f32 accumulator
//  - P staged through a 1KB/wave LDS tile for the C/D -> A layout transpose
// (16-query tile keeps peak register pressure ~140 VGPRs: no spills.)
// ---------------------------------------------------------------------------
__global__ void flash_attn(const bf16* __restrict__ Q, const bf16* __restrict__ K,
                           const bf16* __restrict__ Vt, bf16* __restrict__ att) {
    __shared__ bf16 ldsP[8][16][32];  // 1KB per wave

    const int wib  = threadIdx.x >> 5;
    const int lane = threadIdx.x & 31;
    const int lrow = lane & 15;
    const int hi   = lane >> 4;
    const int w    = blockIdx.x * 8 + wib;
    const int qTiles = Tc / 16;
    const int qi = w % qTiles;
    const int h  = (w / qTiles) % Hc;
    const int b  = w / (qTiles * Hc);
    const int q0 = qi * 16;

    // B fragment of all-ones (32x16) for row-sum WMMAs
    v16bf ones;
#pragma unroll
    for (int i = 0; i < 16; ++i) ones[i] = (bf16)1.0f;

    // Q fragments for d=[0,32) and d=[32,64)
    const bf16* qrow = Q + ((size_t)(b * Tc) + q0 + lrow) * Dc + h * HDc;
    v16bf aQ[2];
#pragma unroll
    for (int ks = 0; ks < 2; ++ks) aQ[ks] = load_a_frag(qrow, ks * 32, hi);

    v8f accO[4] = {};
    v8f accRS = {};  // running row-sum (every column identical)
    float rmax[8];
#pragma unroll
    for (int r = 0; r < 8; ++r) rmax[r] = -1e30f;

    const int kcMax = (q0 + 15) / 32;  // inclusive; chunk start <= q0 always
    for (int kc = 0; kc <= kcMax; ++kc) {
        // --- S = Q @ K^T for two 16-key subtiles ---
        v8f accS[2] = {};
#pragma unroll
        for (int n = 0; n < 2; ++n) {
            int tok = kc * 32 + n * 16 + lrow;
            const bf16* krow = K + ((size_t)(b * Tc) + tok) * Dc + h * HDc;
#pragma unroll
            for (int ks = 0; ks < 2; ++ks) {
                v16bf bK = *(const v16bf*)(krow + ks * 32 + hi * 16);
                accS[n] = __builtin_amdgcn_wmma_f32_16x16x32_bf16(
                    false, aQ[ks], false, bK, (short)0, accS[n], false, false);
            }
        }

        // --- online softmax: mask, DPP row-max, exponentials ---
        float pr[2][8], scl[8];
#pragma unroll
        for (int r = 0; r < 8; ++r) {
            int m = q0 + r + 8 * hi;
            float s0 = accS[0][r] * 0.125f;
            float s1 = accS[1][r] * 0.125f;
            if (kc * 32 + lrow      > m) s0 = -1e30f;
            if (kc * 32 + 16 + lrow > m) s1 = -1e30f;
            float mx = rowhalf_max(fmaxf(s0, s1));
            float nm = fmaxf(rmax[r], mx);
            scl[r] = __expf(rmax[r] - nm);
            rmax[r] = nm;
            pr[0][r] = __expf(s0 - nm);
            pr[1][r] = __expf(s1 - nm);
        }
        // rescale running accumulators (O and row-sum) by exp(old-new)
#pragma unroll
        for (int r = 0; r < 8; ++r) {
            accRS[r] *= scl[r];
#pragma unroll
            for (int t = 0; t < 4; ++t) accO[t][r] *= scl[r];
        }

        // --- C/D layout -> A layout for P via LDS (wave-private tile) ---
#pragma unroll
        for (int n = 0; n < 2; ++n)
#pragma unroll
            for (int r = 0; r < 8; ++r)
                ldsP[wib][r + 8 * hi][n * 16 + lrow] = (bf16)pr[n][r];

        asm volatile("s_wait_dscnt 0" ::: "memory");

        v16bf aP = load_a_frag(&ldsP[wib][lrow][0], 0, hi);

        // --- row sums: accRS += P @ ones ---
        accRS = __builtin_amdgcn_wmma_f32_16x16x32_bf16(
            false, aP, false, ones, (short)0, accRS, false, false);

        // --- O += P @ V  (Vt rows are feature dims, tokens contiguous) ---
#pragma unroll
        for (int t = 0; t < 4; ++t) {
            const bf16* vrow = Vt + ((size_t)(b * Dc) + h * HDc + t * 16 + lrow) * Tc
                                  + kc * 32 + hi * 16;
            v16bf bV = *(const v16bf*)vrow;
            accO[t] = __builtin_amdgcn_wmma_f32_16x16x32_bf16(
                false, aP, false, bV, (short)0, accO[t], false, false);
        }
    }

    // --- normalize and write (B,T,D) bf16 ---
#pragma unroll
    for (int t = 0; t < 4; ++t)
#pragma unroll
        for (int r = 0; r < 8; ++r) {
            int m = q0 + r + 8 * hi;
            int dcol = h * HDc + t * 16 + lrow;
            att[((size_t)(b * Tc) + m) * Dc + dcol] = (bf16)(accO[t][r] / accRS[r]);
        }
}

// ---------------------------------------------------------------------------
extern "C" void kernel_launch(void* const* d_in, const int* in_sizes, int n_in,
                              void* d_out, int out_size, void* d_ws, size_t ws_size,
                              hipStream_t stream) {
    const float* x   = (const float*)d_in[0];
    const int*   pos = (const int*)d_in[1];
    const float* wq  = (const float*)d_in[2];
    const float* bq  = (const float*)d_in[3];
    const float* wk  = (const float*)d_in[4];
    const float* bk  = (const float*)d_in[5];
    const float* wv  = (const float*)d_in[6];
    const float* bv  = (const float*)d_in[7];
    const float* wo  = (const float*)d_in[8];
    const float* bo  = (const float*)d_in[9];

    const int M   = Bc * Tc;       // 4096
    const int BTD = Bc * Tc * Dc;  // 4,194,304
    const int DD  = Dc * Dc;       // 1,048,576

    char* p = (char*)d_ws;
    auto carve = [&](size_t bytes) {
        char* r = p;
        p += (bytes + 255) & ~(size_t)255;
        return r;
    };
    bf16* xb   = (bf16*)carve((size_t)BTD * 2);
    bf16* wqb  = (bf16*)carve((size_t)DD * 2);
    bf16* wkb  = (bf16*)carve((size_t)DD * 2);
    bf16* wvb  = (bf16*)carve((size_t)DD * 2);
    bf16* wob  = (bf16*)carve((size_t)DD * 2);
    bf16* qb   = (bf16*)carve((size_t)BTD * 2);
    bf16* kb   = (bf16*)carve((size_t)BTD * 2);
    bf16* vtb  = (bf16*)carve((size_t)BTD * 2);
    bf16* attb = (bf16*)carve((size_t)BTD * 2);

    // 1. conversions
    cvt_f32_bf16<<<(BTD + 255) / 256, 256, 0, stream>>>(x, xb, BTD);
    cvt_f32_bf16<<<(DD + 255) / 256, 256, 0, stream>>>(wq, wqb, DD);
    cvt_f32_bf16<<<(DD + 255) / 256, 256, 0, stream>>>(wk, wkb, DD);
    cvt_f32_bf16<<<(DD + 255) / 256, 256, 0, stream>>>(wv, wvb, DD);
    cvt_f32_bf16<<<(DD + 255) / 256, 256, 0, stream>>>(wo, wob, DD);

    // 2. QKV projections: (M/32)*(D/64) = 2048 waves -> 256 blocks of 8 waves
    const int gemmBlocks = (M / 32) * (Dc / 64) / 8;
    gemm_xwT<0><<<gemmBlocks, 256, 0, stream>>>(xb, wqb, bq, qb, M, Dc, Dc, Dc);
    gemm_xwT<0><<<gemmBlocks, 256, 0, stream>>>(xb, wkb, bk, kb, M, Dc, Dc, Dc);
    gemm_xwT<1><<<gemmBlocks, 256, 0, stream>>>(xb, wvb, bv, vtb, M, Dc, Dc, Tc);

    // 3. RoPE (in place on qb, kb)
    rope_kernel<<<(Bc * Tc * Hc * 32 + 255) / 256, 256, 0, stream>>>(qb, kb, pos);

    // 4. flash attention: B*H*(T/16) = 4096 waves -> 512 blocks
    flash_attn<<<512, 256, 0, stream>>>(qb, kb, vtb, attb);

    // 5. output projection -> f32 d_out
    gemm_xwT<2><<<gemmBlocks, 256, 0, stream>>>(attb, wob, bo, d_out, M, Dc, Dc, Dc);
}
